// CausalSelfAttention_8993661517916
// MI455X (gfx1250) — matrix operand order repaired and checked by reading, not compiled, not run on hardware
//
#include <hip/hip_runtime.h>
#include <hip/hip_fp16.h>

typedef __attribute__((ext_vector_type(16))) _Float16 v16h;
typedef __attribute__((ext_vector_type(8)))  _Float16 v8h;
typedef __attribute__((ext_vector_type(8)))  float    v8f;

#define T_SEQ   2048
#define BATCH   2
#define C_MODEL 1024
#define NHEAD   16
#define DHEAD   64
#define M_TOT   (BATCH * T_SEQ)   // 4096

static __device__ __forceinline__ v8f wmma_f16(v16h a, v16h b, v8f c) {
  // D = A(16x32 f16) * B(32x16 f16) + C(16x16 f32)
  return __builtin_amdgcn_wmma_f32_16x16x32_f16(false, a, false, b, (short)0, c,
                                                false, false);
}

// ---------------------------------------------------------------------------
// Kernel 1: qkv = x @ W_attn + b_attn, scattered to Q[B,H,T,64], K[B,H,T,64],
//           Vt[B,H,64,T] as f16.  Block tile 128(M) x 128(N), 8 waves,
//           each wave computes 32x64 via 2x4 C-fragments (B frags reused).
// ---------------------------------------------------------------------------
__global__ __launch_bounds__(256) void qkv_gemm_kernel(
    const float* __restrict__ X, const float* __restrict__ W,
    const float* __restrict__ bias, _Float16* __restrict__ Qo,
    _Float16* __restrict__ Ko, _Float16* __restrict__ Vt) {
  __shared__ alignas(64) _Float16 ldsA[128 * 32];
  __shared__ alignas(64) _Float16 ldsB[128 * 32];  // transposed: [n][k]

  const int tid  = threadIdx.x;
  const int lane = tid & 31;
  const int wid  = tid >> 5;      // 0..7
  const int lo   = lane & 15;
  const int hi   = lane >> 4;     // 0 or 1
  const int m0   = blockIdx.x * 128;
  const int n0   = blockIdx.y * 128;
  const int N    = 3 * C_MODEL;

  v8f acc[2][4] = {};

  for (int kt = 0; kt < C_MODEL; kt += 32) {
    __syncthreads();
    {  // stage A: 128x32 f32 -> f16, 16 contiguous elems per thread
      const int row = tid >> 1;
      const int col = (tid & 1) * 16;
      const float* src = X + (size_t)(m0 + row) * C_MODEL + kt + col;
      v8h t0, t1;
#pragma unroll
      for (int i = 0; i < 8; ++i) { t0[i] = (_Float16)src[i]; t1[i] = (_Float16)src[8 + i]; }
      *(v8h*)&ldsA[row * 32 + col]     = t0;
      *(v8h*)&ldsA[row * 32 + col + 8] = t1;
      if (kt + 32 < C_MODEL) __builtin_prefetch(src + 32, 0, 1);
    }
    {  // stage B transposed: ldsB[n][k], 16 strided elems per thread
      const int n  = tid & 127;
      const int kk = (tid >> 7) * 16;
      const float* src = W + (size_t)(kt + kk) * N + n0 + n;
#pragma unroll
      for (int i = 0; i < 16; ++i)
        ldsB[n * 32 + kk + i] = (_Float16)src[(size_t)i * N];
      if (kt + 32 < C_MODEL) __builtin_prefetch(src + (size_t)32 * N, 0, 1);
    }
    __syncthreads();

    // two A fragments for this wave's 32-row slice
    const int ar0 = 32 * (wid >> 1);
    v16h a[2];
#pragma unroll
    for (int mi = 0; mi < 2; ++mi) {
      const int arow = ar0 + 16 * mi + lo;
      const v8h alo = *(const v8h*)&ldsA[arow * 32 + 16 * hi];
      const v8h ahi = *(const v8h*)&ldsA[arow * 32 + 16 * hi + 8];
#pragma unroll
      for (int i = 0; i < 8; ++i) { a[mi][i] = alo[i]; a[mi][i + 8] = ahi[i]; }
    }
#pragma unroll
    for (int j = 0; j < 4; ++j) {
      const int bcol = 64 * (wid & 1) + 16 * j + lo;
      const v16h b = *(const v16h*)&ldsB[bcol * 32 + 16 * hi];
      acc[0][j] = wmma_f16(a[0], b, acc[0][j]);
      acc[1][j] = wmma_f16(a[1], b, acc[1][j]);
    }
  }

  // epilogue: + bias, route to Q / K / V^T
#pragma unroll
  for (int j = 0; j < 4; ++j) {
    const int n      = n0 + 64 * (wid & 1) + 16 * j + lo;
    const float bv   = bias[n];
    const int which  = n >> 10;        // 0=q 1=k 2=v (wave-uniform per j)
    const int within = n & 1023;
    const int h      = within >> 6;
    const int dd     = within & 63;
#pragma unroll
    for (int mi = 0; mi < 2; ++mi)
#pragma unroll
      for (int r = 0; r < 8; ++r) {
        const int m  = m0 + 32 * (wid >> 1) + 16 * mi + r + 8 * hi;
        const int bb = m >> 11;        // T_SEQ == 2048
        const int t  = m & (T_SEQ - 1);
        const _Float16 hv = (_Float16)(acc[mi][j][r] + bv);
        if (which == 0)
          Qo[((size_t)(bb * NHEAD + h) * T_SEQ + t) * DHEAD + dd] = hv;
        else if (which == 1)
          Ko[((size_t)(bb * NHEAD + h) * T_SEQ + t) * DHEAD + dd] = hv;
        else
          Vt[((size_t)(bb * NHEAD + h) * DHEAD + dd) * T_SEQ + t] = hv;
      }
  }
}

// ---------------------------------------------------------------------------
// Kernel 2: flash-style causal attention.  grid = (T/128, B*H), 4 waves/block,
// each wave owns a 32-row query tile, iterates keys 32 at a time; K/V
// fragments are reused across both 16-row sub-tiles.
// ---------------------------------------------------------------------------
__global__ __launch_bounds__(128) void attn_kernel(
    const _Float16* __restrict__ Q, const _Float16* __restrict__ K,
    const _Float16* __restrict__ Vt, const int* __restrict__ amask,
    _Float16* __restrict__ Yatt) {
  __shared__ alignas(64) _Float16 Pbuf[4][32 * 32];

  const int tid  = threadIdx.x;
  const int lane = tid & 31;
  const int wid  = tid >> 5;  // 0..3
  const int lo   = lane & 15;
  const int hi   = lane >> 4;
  const int bh   = blockIdx.y;      // 0..31
  const int b    = bh >> 4;
  const int h    = bh & 15;
  const int q0   = blockIdx.x * 128 + wid * 32;   // multiple of 32
  const float NEG = -__builtin_inff();

  // Load Q fragments (A operand, K-dim = head dim), pre-scaled by 1/sqrt(64)
  v16h aq[2][2];  // [m-subtile][d-fragment]
#pragma unroll
  for (int mi = 0; mi < 2; ++mi) {
    const _Float16* qbase = Q + ((size_t)bh * T_SEQ + q0 + 16 * mi + lo) * DHEAD;
#pragma unroll
    for (int f = 0; f < 2; ++f) {
      const v8h l0 = *(const v8h*)(qbase + 32 * f + 16 * hi);
      const v8h l1 = *(const v8h*)(qbase + 32 * f + 16 * hi + 8);
#pragma unroll
      for (int i = 0; i < 8; ++i) {
        aq[mi][f][i]     = l0[i] * (_Float16)0.125f;
        aq[mi][f][i + 8] = l1[i] * (_Float16)0.125f;
      }
    }
  }

  v8f o[2][4] = {};
  float m8[2][8], l8[2][8];
#pragma unroll
  for (int mi = 0; mi < 2; ++mi)
#pragma unroll
    for (int r = 0; r < 8; ++r) { m8[mi][r] = NEG; l8[mi][r] = 0.f; }

  const int nIter = (q0 + 32) >> 5;  // keys [0, q0+32), wave-uniform
  const _Float16* kbase = K + (size_t)bh * T_SEQ * DHEAD;
  const _Float16* vbase = Vt + (size_t)bh * DHEAD * T_SEQ;
  const int* mrow = amask + b * T_SEQ;
  _Float16* P = &Pbuf[wid][0];

  for (int it = 0; it < nIter; ++it) {
    const int kb = it * 32;
    // ---- scores for two 16-key tiles: S = Q * K^T (K frags reused) ----
    v8f s[2][2];  // [m-subtile][key-tile]
#pragma unroll
    for (int t2 = 0; t2 < 2; ++t2) {
      const int kcol = kb + 16 * t2 + lo;
      const _Float16* kp = kbase + (size_t)kcol * DHEAD;
      const v16h bk0 = *(const v16h*)(kp + 16 * hi);
      const v16h bk1 = *(const v16h*)(kp + 32 + 16 * hi);
      const int km = mrow[kcol];
#pragma unroll
      for (int mi = 0; mi < 2; ++mi) {
        v8f sc = {};
        sc = wmma_f16(aq[mi][0], bk0, sc);
        sc = wmma_f16(aq[mi][1], bk1, sc);
#pragma unroll
        for (int r = 0; r < 8; ++r) {
          const int qrow = q0 + 16 * mi + r + 8 * hi;
          const bool keep = (kcol <= qrow) && (km != 0);
          sc[r] = keep ? sc[r] : NEG;
        }
        s[mi][t2] = sc;
      }
    }
    // ---- online softmax (row reductions within 16-lane groups) ----
    float corr[2][8];
#pragma unroll
    for (int mi = 0; mi < 2; ++mi)
#pragma unroll
      for (int r = 0; r < 8; ++r) {
        float mx = fmaxf(s[mi][0][r], s[mi][1][r]);
        mx = fmaxf(mx, __shfl_xor(mx, 1));
        mx = fmaxf(mx, __shfl_xor(mx, 2));
        mx = fmaxf(mx, __shfl_xor(mx, 4));
        mx = fmaxf(mx, __shfl_xor(mx, 8));
        const float mnew = fmaxf(m8[mi][r], mx);
        const float c = (m8[mi][r] == NEG) ? 0.f : __expf(m8[mi][r] - mnew);
        corr[mi][r] = c;
        const float p0 = (s[mi][0][r] == NEG) ? 0.f : __expf(s[mi][0][r] - mnew);
        const float p1 = (s[mi][1][r] == NEG) ? 0.f : __expf(s[mi][1][r] - mnew);
        float rs = p0 + p1;
        rs += __shfl_xor(rs, 1);
        rs += __shfl_xor(rs, 2);
        rs += __shfl_xor(rs, 4);
        rs += __shfl_xor(rs, 8);
        l8[mi][r] = l8[mi][r] * c + rs;
        m8[mi][r] = mnew;
        const int prow = 16 * mi + r + 8 * hi;  // C-layout -> [row][col] in LDS
        P[prow * 32 + lo]      = (_Float16)p0;
        P[prow * 32 + 16 + lo] = (_Float16)p1;
      }
    asm volatile("s_wait_dscnt 0x0" ::: "memory");  // cross-lane LDS handoff
#pragma unroll
    for (int mi = 0; mi < 2; ++mi)
#pragma unroll
      for (int j = 0; j < 4; ++j)
#pragma unroll
        for (int r = 0; r < 8; ++r) o[mi][j][r] *= corr[mi][r];

    // reload P in A-fragment layout (one per m-subtile)
    v16h ap[2];
#pragma unroll
    for (int mi = 0; mi < 2; ++mi) {
      const v8h plo = *(const v8h*)&P[(16 * mi + lo) * 32 + 16 * hi];
      const v8h phi = *(const v8h*)&P[(16 * mi + lo) * 32 + 16 * hi + 8];
#pragma unroll
      for (int i = 0; i < 8; ++i) { ap[mi][i] = plo[i]; ap[mi][i + 8] = phi[i]; }
    }
    // ---- O += P * V  (B operand from V^T, reused across m-subtiles) ----
#pragma unroll
    for (int j = 0; j < 4; ++j) {
      const _Float16* vp = vbase + (size_t)(16 * j + lo) * T_SEQ + kb + 16 * hi;
      const v16h bv = *(const v16h*)vp;
      o[0][j] = wmma_f16(ap[0], bv, o[0][j]);
      o[1][j] = wmma_f16(ap[1], bv, o[1][j]);
    }
  }

  // ---- finalize: 1/l, query-mask zeroing, store f16 [B,T,C] ----
  _Float16* ybase = Yatt + (size_t)b * T_SEQ * C_MODEL + h * DHEAD;
#pragma unroll
  for (int mi = 0; mi < 2; ++mi) {
    float inv[8];
#pragma unroll
    for (int r = 0; r < 8; ++r) {
      const int qrow = q0 + 16 * mi + r + 8 * hi;
      const int qm = mrow[qrow];
      inv[r] = (qm != 0 && l8[mi][r] > 0.f) ? (1.0f / l8[mi][r]) : 0.f;
    }
#pragma unroll
    for (int j = 0; j < 4; ++j)
#pragma unroll
      for (int r = 0; r < 8; ++r) {
        const int qrow = q0 + 16 * mi + r + 8 * hi;
        ybase[(size_t)qrow * C_MODEL + 16 * j + lo] =
            (_Float16)(o[mi][j][r] * inv[r]);
      }
  }
}

// ---------------------------------------------------------------------------
// Kernel 3: out = Yatt(f16) @ W_proj + b_proj   (f32 output)
// ---------------------------------------------------------------------------
__global__ __launch_bounds__(256) void proj_gemm_kernel(
    const _Float16* __restrict__ Y, const float* __restrict__ W,
    const float* __restrict__ bias, float* __restrict__ Out) {
  __shared__ alignas(64) _Float16 ldsA[128 * 32];
  __shared__ alignas(64) _Float16 ldsB[128 * 32];

  const int tid  = threadIdx.x;
  const int lane = tid & 31;
  const int wid  = tid >> 5;
  const int lo   = lane & 15;
  const int hi   = lane >> 4;
  const int m0   = blockIdx.x * 128;
  const int n0   = blockIdx.y * 128;
  const int N    = C_MODEL;

  v8f acc[2][4] = {};

  for (int kt = 0; kt < C_MODEL; kt += 32) {
    __syncthreads();
    {  // stage A (already f16): 16 contiguous halves per thread
      const int row = tid >> 1;
      const int col = (tid & 1) * 16;
      const _Float16* src = Y + (size_t)(m0 + row) * C_MODEL + kt + col;
      *(v8h*)&ldsA[row * 32 + col]     = *(const v8h*)src;
      *(v8h*)&ldsA[row * 32 + col + 8] = *(const v8h*)(src + 8);
      if (kt + 32 < C_MODEL) __builtin_prefetch(src + 32, 0, 1);
    }
    {  // stage B transposed
      const int n  = tid & 127;
      const int kk = (tid >> 7) * 16;
      const float* src = W + (size_t)(kt + kk) * N + n0 + n;
#pragma unroll
      for (int i = 0; i < 16; ++i)
        ldsB[n * 32 + kk + i] = (_Float16)src[(size_t)i * N];
      if (kt + 32 < C_MODEL) __builtin_prefetch(src + (size_t)32 * N, 0, 1);
    }
    __syncthreads();

    const int ar0 = 32 * (wid >> 1);
    v16h a[2];
#pragma unroll
    for (int mi = 0; mi < 2; ++mi) {
      const int arow = ar0 + 16 * mi + lo;
      const v8h alo = *(const v8h*)&ldsA[arow * 32 + 16 * hi];
      const v8h ahi = *(const v8h*)&ldsA[arow * 32 + 16 * hi + 8];
#pragma unroll
      for (int i = 0; i < 8; ++i) { a[mi][i] = alo[i]; a[mi][i + 8] = ahi[i]; }
    }
#pragma unroll
    for (int j = 0; j < 4; ++j) {
      const int bcol = 64 * (wid & 1) + 16 * j + lo;
      const v16h bfrag = *(const v16h*)&ldsB[bcol * 32 + 16 * hi];
      acc[0][j] = wmma_f16(a[0], bfrag, acc[0][j]);
      acc[1][j] = wmma_f16(a[1], bfrag, acc[1][j]);
    }
  }

#pragma unroll
  for (int j = 0; j < 4; ++j) {
    const int n = n0 + 64 * (wid & 1) + 16 * j + lo;
    const float bv = bias[n];
#pragma unroll
    for (int mi = 0; mi < 2; ++mi)
#pragma unroll
      for (int r = 0; r < 8; ++r) {
        const int m = m0 + 32 * (wid >> 1) + 16 * mi + r + 8 * hi;
        Out[(size_t)m * C_MODEL + n] = acc[mi][j][r] + bv;
      }
  }
}

// ---------------------------------------------------------------------------
extern "C" void kernel_launch(void* const* d_in, const int* in_sizes, int n_in,
                              void* d_out, int out_size, void* d_ws,
                              size_t ws_size, hipStream_t stream) {
  const float* x      = (const float*)d_in[0];
  const int*   amask  = (const int*)d_in[1];
  const float* W_attn = (const float*)d_in[2];
  const float* b_attn = (const float*)d_in[3];
  const float* W_proj = (const float*)d_in[4];
  const float* b_proj = (const float*)d_in[5];
  float* out = (float*)d_out;

  const size_t perTensor = (size_t)BATCH * NHEAD * T_SEQ * DHEAD;  // 4M halves
  _Float16* Qf = (_Float16*)d_ws;
  _Float16* Kf = Qf + perTensor;
  _Float16* Vt = Kf + perTensor;
  _Float16* Ya = Vt + perTensor;  // [B,T,C] f16; total ws use = 32 MB

  dim3 g1(M_TOT / 128, (3 * C_MODEL) / 128);
  qkv_gemm_kernel<<<g1, 256, 0, stream>>>(x, W_attn, b_attn, Qf, Kf, Vt);

  dim3 g2(T_SEQ / 128, BATCH * NHEAD);
  attn_kernel<<<g2, 128, 0, stream>>>(Qf, Kf, Vt, amask, Ya);

  dim3 g3(M_TOT / 128, C_MODEL / 128);
  proj_gemm_kernel<<<g3, 256, 0, stream>>>(Ya, W_proj, b_proj, out);
}
